// CausalSelfAttention_10505490006306
// MI455X (gfx1250) — compile-verified, matrix operand order checked by reading, not run on hardware
//
#include <hip/hip_runtime.h>

// ---------- types ----------
typedef _Float16 v16h __attribute__((ext_vector_type(16)));
typedef _Float16 h8   __attribute__((ext_vector_type(8)));
typedef float    v8f  __attribute__((ext_vector_type(8)));

#define HEADS 16
#define TSEQ  2048
#define CDIM  1024
#define DHEAD 64

// ---------- WMMA helpers ----------
__device__ __forceinline__ v8f wmma16(v16h a, v16h b, v8f c) {
  // D = A(16x32 f16) * B(32x16 f16) + C(16x16 f32)
  return __builtin_amdgcn_wmma_f32_16x16x32_f16(
      /*neg_a=*/false, a, /*neg_b=*/false, b,
      /*c_mod=*/(short)0, c, /*reuse_a=*/false, /*reuse_b=*/false);
}

// Load one A/B operand fragment (16x32 halves) for this lane.
// p = &Mat[row_for_this_lane][k_base]; per ISA 7.12.2 the lane's halves are
// two contiguous 8-half (16B) runs at k offsets hi*8 and 16+hi*8.
__device__ __forceinline__ v16h load_frag16(const _Float16* p, int hi) {
  union { v16h v; h8 h[2]; } u;
  u.h[0] = *(const h8*)(p + hi * 8);
  u.h[1] = *(const h8*)(p + 16 + hi * 8);
  return u.v;
}

// ---------- conversion kernels ----------
__global__ void k_f32_to_f16(const float* __restrict__ in,
                             _Float16* __restrict__ out, int n) {
  int i = blockIdx.x * blockDim.x + threadIdx.x;
  if (i < n) out[i] = (_Float16)in[i];
}

// LDS-tiled transpose+convert: in [R(k) x Ccols(n)] f32 -> out [Ccols][R] f16.
// Both global sides coalesced. Block (32,8), 32x32 tile.
__global__ __launch_bounds__(256) void k_transpose_f16(
    const float* __restrict__ in, _Float16* __restrict__ out, int R, int Ccols) {
  __shared__ _Float16 tile[32][33];
  const int tx = threadIdx.x, ty = threadIdx.y;
  const int r0 = blockIdx.y * 32, c0 = blockIdx.x * 32;
#pragma unroll
  for (int i = ty; i < 32; i += 8)
    tile[i][tx] = (_Float16)in[(size_t)(r0 + i) * Ccols + c0 + tx];
  __syncthreads();
#pragma unroll
  for (int i = ty; i < 32; i += 8)
    out[(size_t)(c0 + i) * R + r0 + tx] = tile[tx][i];
}

// ---------- shared WMMA GEMM core ----------
// C_blk[128x256] += A[m_blk..+128, 0..1024) * B^T rows (Bt[n][k], n_blk..+256).
// Double-buffered LDS staging (A: 2x8KB, B: 2x16KB), 8 waves = 2(m) x 4(n),
// each wave accumulates a 64x64 tile (16 WMMA accumulators).
__device__ __forceinline__ void wmma_gemm_128x256(
    const _Float16* __restrict__ A, const _Float16* __restrict__ Bt,
    int m_blk, int n_blk, _Float16* As, _Float16* Bs, v8f (&acc)[4][4]) {
  const int tid = threadIdx.x, lane = tid & 31, w = tid >> 5;
  const int lo = lane & 15, hi = lane >> 4;
  const int wm = w & 1, wn = w >> 1;
  const int sr = tid >> 2;          // staging row 0..63
  const int sk = (tid & 3) * 8;     // staging k offset 0/8/16/24

  h8 ra[2], rb[4];
#pragma unroll
  for (int q = 0; q < 2; ++q)
    ra[q] = *(const h8*)(A + (size_t)(m_blk + sr + q * 64) * CDIM + sk);
#pragma unroll
  for (int q = 0; q < 4; ++q)
    rb[q] = *(const h8*)(Bt + (size_t)(n_blk + sr + q * 64) * CDIM + sk);
#pragma unroll
  for (int q = 0; q < 2; ++q)
    *(h8*)(As + (sr + q * 64) * 32 + sk) = ra[q];
#pragma unroll
  for (int q = 0; q < 4; ++q)
    *(h8*)(Bs + (sr + q * 64) * 32 + sk) = rb[q];
  __syncthreads();

  int buf = 0;
  const int NT = CDIM / 32;         // 32 K-steps
  for (int t = 0; t < NT; ++t) {
    const int kbn = (t + 1) * 32;
    if (t + 1 < NT) {               // fetch next tile into registers
#pragma unroll
      for (int q = 0; q < 2; ++q)
        ra[q] = *(const h8*)(A + (size_t)(m_blk + sr + q * 64) * CDIM + kbn + sk);
#pragma unroll
      for (int q = 0; q < 4; ++q)
        rb[q] = *(const h8*)(Bt + (size_t)(n_blk + sr + q * 64) * CDIM + kbn + sk);
    }
    if (t + 2 < NT) {               // pull tile t+2 toward the WGP
      const int kb2 = (t + 2) * 32;
      if (tid < 128)
        __builtin_prefetch(A + (size_t)(m_blk + tid) * CDIM + kb2, 0, 0);
      __builtin_prefetch(Bt + (size_t)(n_blk + tid) * CDIM + kb2, 0, 0);
    }
    // compute from LDS[buf] (hides the global loads above)
    const _Float16* Ab = As + buf * (128 * 32);
    const _Float16* Bb = Bs + buf * (256 * 32);
    v16h a[4], b[4];
#pragma unroll
    for (int i = 0; i < 4; ++i)
      a[i] = load_frag16(Ab + (wm * 64 + i * 16 + lo) * 32, hi);
#pragma unroll
    for (int j = 0; j < 4; ++j)
      b[j] = load_frag16(Bb + (wn * 64 + j * 16 + lo) * 32, hi);
#pragma unroll
    for (int i = 0; i < 4; ++i)
#pragma unroll
      for (int j = 0; j < 4; ++j) acc[i][j] = wmma16(a[i], b[j], acc[i][j]);
    if (t + 1 < NT) {               // stage next tile into the other buffer
      _Float16* An = As + (buf ^ 1) * (128 * 32);
      _Float16* Bn = Bs + (buf ^ 1) * (256 * 32);
#pragma unroll
      for (int q = 0; q < 2; ++q) *(h8*)(An + (sr + q * 64) * 32 + sk) = ra[q];
#pragma unroll
      for (int q = 0; q < 4; ++q) *(h8*)(Bn + (sr + q * 64) * 32 + sk) = rb[q];
    }
    __syncthreads();
    buf ^= 1;
  }
}

// ---------- QKV GEMM + bias + head scatter ----------
// Q[B,H,T,D] (scaled 1/8), K[B,H,T,D], Vt[B,H,D,T], all f16.
__global__ __launch_bounds__(256) void k_qkv_gemm(
    const _Float16* __restrict__ xh, const _Float16* __restrict__ Wt,
    const float* __restrict__ bias, _Float16* __restrict__ Q,
    _Float16* __restrict__ K, _Float16* __restrict__ Vt) {
  __shared__ _Float16 As[2 * 128 * 32];
  __shared__ _Float16 Bs[2 * 256 * 32];
  const int tid = threadIdx.x, lane = tid & 31, w = tid >> 5;
  const int lo = lane & 15, hi = lane >> 4;
  const int m_blk = blockIdx.y * 128, n_blk = blockIdx.x * 256;
  const int m0 = m_blk + (w & 1) * 64;
  const int n0 = n_blk + (w >> 1) * 64;

  v8f acc[4][4] = {};
  wmma_gemm_128x256(xh, Wt, m_blk, n_blk, As, Bs, acc);

#pragma unroll
  for (int i = 0; i < 4; ++i) {
    const int tbase = m0 + i * 16 + hi * 8;        // 8 consecutive tokens
    const int bb = tbase >> 11, t0 = tbase & (TSEQ - 1);
#pragma unroll
    for (int j = 0; j < 4; ++j) {
      const int col16 = n0 + j * 16;               // uniform per fragment
      const int col = col16 + lo;
      const float bj = bias[col];
      if (col16 >= 2 * CDIM) {                     // V region: vector store
        const int c2 = col - 2 * CDIM, hv = c2 >> 6, d = c2 & 63;
        h8 vv;
#pragma unroll
        for (int r = 0; r < 8; ++r) vv[r] = (_Float16)(acc[i][j][r] + bj);
        *(h8*)(Vt + (((size_t)bb * HEADS + hv) * DHEAD + d) * TSEQ + t0) = vv;
      } else if (col16 >= CDIM) {                  // K
        const int c2 = col - CDIM, hk = c2 >> 6, d = c2 & 63;
#pragma unroll
        for (int r = 0; r < 8; ++r)
          K[(((size_t)bb * HEADS + hk) * TSEQ + t0 + r) * DHEAD + d] =
              (_Float16)(acc[i][j][r] + bj);
      } else {                                     // Q, fold 1/sqrt(64)
        const int hq = col >> 6, d = col & 63;
#pragma unroll
        for (int r = 0; r < 8; ++r)
          Q[(((size_t)bb * HEADS + hq) * TSEQ + t0 + r) * DHEAD + d] =
              (_Float16)((acc[i][j][r] + bj) * 0.125f);
      }
    }
  }
}

// ---------- causal flash attention ----------
// grid (T/64, H, B); 4 waves/block, each wave owns 16 query rows.
__global__ __launch_bounds__(128) void k_attn(
    const _Float16* __restrict__ Q, const _Float16* __restrict__ K,
    const _Float16* __restrict__ Vt, _Float16* __restrict__ yh) {
  __shared__ _Float16 Plds[4 * 16 * 32];           // per-wave 16x32 P tile
  const int tid = threadIdx.x, lane = tid & 31, w = tid >> 5;
  const int lo = lane & 15, hi = lane >> 4;
  const int b = blockIdx.z, h = blockIdx.y;
  const int qb = blockIdx.x * 64 + w * 16;

  const _Float16* Qh = Q + ((size_t)b * HEADS + h) * TSEQ * DHEAD;
  const _Float16* Kh = K + ((size_t)b * HEADS + h) * TSEQ * DHEAD;
  const _Float16* Vh = Vt + ((size_t)b * HEADS + h) * DHEAD * TSEQ;
  _Float16* myP = Plds + w * (16 * 32);

  const v16h aq0 = load_frag16(Qh + (size_t)(qb + lo) * DHEAD + 0, hi);
  const v16h aq1 = load_frag16(Qh + (size_t)(qb + lo) * DHEAD + 32, hi);

  v8f o[4] = {};
  float mrow[8], lrow[8];
#pragma unroll
  for (int r = 0; r < 8; ++r) { mrow[r] = -1e30f; lrow[r] = 0.f; }

  const int ktmax = (qb + 16 + 31) >> 5;           // cover keys <= qb+15
  for (int kt = 0; kt < ktmax; ++kt) {
    const int k0 = kt * 32;
    if (kt + 1 < ktmax) {                          // prefetch next K/V tile
      const int kn = k0 + 32;
      __builtin_prefetch(Kh + (size_t)(kn + lane) * DHEAD, 0, 0);
      __builtin_prefetch(Vh + (size_t)lane * TSEQ + kn, 0, 0);
      __builtin_prefetch(Vh + (size_t)(32 + lane) * TSEQ + kn, 0, 0);
    }
    v8f s[2] = {};
#pragma unroll
    for (int j = 0; j < 2; ++j) {
      const _Float16* kp = Kh + (size_t)(k0 + j * 16 + lo) * DHEAD;
      s[j] = wmma16(aq0, load_frag16(kp + 0, hi), s[j]);
      s[j] = wmma16(aq1, load_frag16(kp + 32, hi), s[j]);
    }
    const bool diag = (k0 + 31) > qb;              // tile crosses diagonal?
#pragma unroll
    for (int r = 0; r < 8; ++r) {
      const int qrow = qb + hi * 8 + r;
      if (diag) {
#pragma unroll
        for (int j = 0; j < 2; ++j)
          if (k0 + j * 16 + lo > qrow) s[j][r] = -1e30f;
      }
      float cm = fmaxf(s[0][r], s[1][r]);
#pragma unroll
      for (int msk = 1; msk <= 8; msk <<= 1)
        cm = fmaxf(cm, __shfl_xor(cm, msk, 32));
      const float mnew = fmaxf(mrow[r], cm);
      const float alpha = __expf(mrow[r] - mnew);
      const float p0 = __expf(s[0][r] - mnew);
      const float p1 = __expf(s[1][r] - mnew);
      float ps = p0 + p1;
#pragma unroll
      for (int msk = 1; msk <= 8; msk <<= 1) ps += __shfl_xor(ps, msk, 32);
      lrow[r] = lrow[r] * alpha + ps;
      mrow[r] = mnew;
#pragma unroll
      for (int jj = 0; jj < 4; ++jj) o[jj][r] *= alpha;
      const int prow = hi * 8 + r;
      myP[prow * 32 + lo] = (_Float16)p0;          // C-layout -> LDS
      myP[prow * 32 + 16 + lo] = (_Float16)p1;
    }
    // reload P in A-layout (same-wave LDS RAW is in-order)
    const v16h ap = load_frag16(myP + lo * 32, hi);
#pragma unroll
    for (int jj = 0; jj < 4; ++jj) {
      const v16h bv = load_frag16(Vh + (size_t)(jj * 16 + lo) * TSEQ + k0, hi);
      o[jj] = wmma16(ap, bv, o[jj]);
    }
  }

#pragma unroll
  for (int jj = 0; jj < 4; ++jj)
#pragma unroll
    for (int r = 0; r < 8; ++r) {
      const int qrow = qb + hi * 8 + r;
      const size_t row = (size_t)b * TSEQ + qrow;
      yh[row * CDIM + h * DHEAD + jj * 16 + lo] =
          (_Float16)(o[jj][r] / lrow[r]);
    }
}

// ---------- projection GEMM: out = yh[8192,1024] @ W_proj + b (f32 out) ----------
__global__ __launch_bounds__(256) void k_proj_gemm(
    const _Float16* __restrict__ yh, const _Float16* __restrict__ Wt,
    const float* __restrict__ bias, float* __restrict__ out) {
  __shared__ _Float16 As[2 * 128 * 32];
  __shared__ _Float16 Bs[2 * 256 * 32];
  const int tid = threadIdx.x, lane = tid & 31, w = tid >> 5;
  const int lo = lane & 15, hi = lane >> 4;
  const int m_blk = blockIdx.y * 128, n_blk = blockIdx.x * 256;
  const int m0 = m_blk + (w & 1) * 64;
  const int n0 = n_blk + (w >> 1) * 64;

  v8f acc[4][4] = {};
  wmma_gemm_128x256(yh, Wt, m_blk, n_blk, As, Bs, acc);

#pragma unroll
  for (int i = 0; i < 4; ++i)
#pragma unroll
    for (int j = 0; j < 4; ++j) {
      const int col = n0 + j * 16 + lo;
      const float bj = bias[col];
#pragma unroll
      for (int r = 0; r < 8; ++r) {
        const int row = m0 + i * 16 + hi * 8 + r;
        out[(size_t)row * CDIM + col] = acc[i][j][r] + bj;
      }
    }
}

// ---------- launcher ----------
extern "C" void kernel_launch(void* const* d_in, const int* in_sizes, int n_in,
                              void* d_out, int out_size, void* d_ws,
                              size_t ws_size, hipStream_t stream) {
  const float* x      = (const float*)d_in[0];   // [4,2048,1024]
  const float* W_attn = (const float*)d_in[1];   // [1024,3072]
  const float* b_attn = (const float*)d_in[2];   // [3072]
  const float* W_proj = (const float*)d_in[3];   // [1024,1024]
  const float* b_proj = (const float*)d_in[4];   // [1024]
  float* out = (float*)d_out;

  const size_t NTOK = 4 * (size_t)TSEQ;          // 8192
  const size_t NX   = NTOK * CDIM;               // 8M elements

  _Float16* ws = (_Float16*)d_ws;
  _Float16* xh     = ws;                        // 8M
  _Float16* WattnT = xh + NX;                   // 3M  [3072][1024]
  _Float16* WprojT = WattnT + 3u * 1024 * 1024; // 1M  [1024][1024]
  _Float16* Qh     = WprojT + 1u * 1024 * 1024; // 8M  [B,H,T,D]
  _Float16* Kh     = Qh + NX;                   // 8M
  _Float16* Vt     = Kh + NX;                   // 8M  [B,H,D,T]
  _Float16* yh     = Vt + NX;                   // 8M
  if (ws_size < 44ull * 1024 * 1024 * 2) return; // 88MB needed

  k_f32_to_f16<<<dim3((NX + 255) / 256), dim3(256), 0, stream>>>(x, xh, (int)NX);
  k_transpose_f16<<<dim3(3 * CDIM / 32, CDIM / 32), dim3(32, 8), 0, stream>>>(
      W_attn, WattnT, CDIM, 3 * CDIM);
  k_transpose_f16<<<dim3(CDIM / 32, CDIM / 32), dim3(32, 8), 0, stream>>>(
      W_proj, WprojT, CDIM, CDIM);

  k_qkv_gemm<<<dim3(3 * CDIM / 256, NTOK / 128), dim3(256), 0, stream>>>(
      xh, WattnT, b_attn, Qh, Kh, Vt);

  k_attn<<<dim3(TSEQ / 64, HEADS, 4), dim3(128), 0, stream>>>(Qh, Kh, Vt, yh);

  k_proj_gemm<<<dim3(CDIM / 256, NTOK / 128), dim3(256), 0, stream>>>(
      yh, WprojT, b_proj, out);
}
